// GeoPointMatching_65944927863072
// MI455X (gfx1250) — compile-verified, matrix operand order checked by reading, not run on hardware
//
#include <hip/hip_runtime.h>
#include <hip/hip_bf16.h>
#include <math.h>

// ---------------------------------------------------------------------------
// GeoPointMatching forward for MI455X (gfx1250, wave32, WMMA 16x16x32 f16)
// LDS tiles are stored pre-swizzled in WMMA fragment order so every operand
// fragment is two contiguous ds_load_b128 per lane (no scalar u16 gathers).
// ---------------------------------------------------------------------------

typedef __attribute__((ext_vector_type(16))) _Float16 v16h;
typedef __attribute__((ext_vector_type(8)))  _Float16 v8h;
typedef __attribute__((ext_vector_type(2)))  _Float16 h2;
typedef __attribute__((ext_vector_type(8)))  float    v8f;

#define CHID 128
#define CNT  197      // sparse tokens incl. background
#define CNP  196      // fps samples
#define CPP  (CNT*CNT)
#define CHEADS 4
#define DENSE_N 4096

#define APITCH 40     // halves per 32-K row (80B, 16B aligned)
#define BPITCH 40
#define GPITCH 136    // halves per 128-K row in fused geo kernels (272B)

// ---------------------------------------------------------------------------
// WMMA helpers (layouts per CDNA5 ISA §7.12.2)
// ---------------------------------------------------------------------------
__device__ __forceinline__ v8f wmma_f16(v16h a, v16h b, v8f c) {
    return __builtin_amdgcn_wmma_f32_16x16x32_f16(
        /*neg_a=*/false, a, /*neg_b=*/false, b,
        /*c_mod=*/(short)0, c, /*reuse_a=*/false, /*reuse_b=*/false);
}

// K-group permutation: store groups-of-8 halves as [g0, g2, g1, g3] so each
// half-wave's required {K0-7,K16-23} / {K8-15,K24-31} set is contiguous.
__device__ __forceinline__ int kperm(int kk) {
    int g = kk >> 3;
    g = ((g << 1) | (g >> 1)) & 3;
    return (g << 3) | (kk & 7);
}

// Load one 16-half fragment: lanes 0-15 read halves [0,16), lanes 16-31 read
// halves [16,32) of a pre-swizzled row.  base must be 16B aligned.
__device__ __forceinline__ v16h load_frag(const _Float16* base, int lane) {
    const _Float16* p = base + ((lane >> 4) << 4);
    v8h lo = *(const v8h*)(p);
    v8h hi = *(const v8h*)(p + 8);
    return __builtin_shufflevector(lo, hi, 0, 1, 2, 3, 4, 5, 6, 7,
                                   8, 9, 10, 11, 12, 13, 14, 15);
}

// ---------------------------------------------------------------------------
// Generic tiled WMMA GEMM:  C = act(op(A)[M,K] * op(B)[K,N] (+bias) (+res))
// Block = 256 threads = 8 waves.  Tile M=128 (wave=16 rows), N=64, K step 32.
// Batched via grid.z; z -> (b = z/Hn, h = z%Hn) with independent strides.
// ---------------------------------------------------------------------------
template <int ACT, int TRA, int TRB, int BIAS, int RES>
__global__ __launch_bounds__(256) void k_gemm(
    const float* __restrict__ A, const float* __restrict__ Bw,
    const float* __restrict__ bias, const float* __restrict__ res,
    float* __restrict__ C, int M, int N, int K, int lda, int ldb, int ldc,
    long long sAb, long long sAh, long long sBb, long long sBh,
    long long sCb, long long sCh, int Hn)
{
    __shared__ __align__(16) _Float16 As[128 * APITCH]; // rows x swizzled K
    __shared__ __align__(16) _Float16 Bs[64 * BPITCH];  // cols x swizzled K

    const int z  = blockIdx.z;
    const int bb = z / Hn, hh = z % Hn;
    A  += (long long)bb * sAb + (long long)hh * sAh;
    Bw += (long long)bb * sBb + (long long)hh * sBh;
    C  += (long long)bb * sCb + (long long)hh * sCh;
    const float* R = RES ? (res + (long long)bb * sCb + (long long)hh * sCh)
                         : nullptr;

    const int m0 = blockIdx.x * 128, n0 = blockIdx.y * 64;
    const int tid = threadIdx.x, lane = tid & 31, wid = tid >> 5;

    v8f acc[4] = {};

    for (int kc = 0; kc < K; kc += 32) {
        if (kc + 32 < K && tid == 0) {
            __builtin_prefetch((const void*)(Bw + (long long)(kc + 32) * ldb), 0, 0);
        }
        // stage A tile 128x32: pair-packed b32 LDS stores at permuted K
#pragma unroll
        for (int i = 0; i < 8; ++i) {
            int idx = tid + (i << 8);           // 2048 K-pairs
            int r   = idx >> 4;                 // 16 pairs per row
            int kp  = (idx & 15) << 1;          // even kk
            int gr = m0 + r, gk = kc + kp;
            float v0 = 0.f, v1 = 0.f;
            if (gr < M) {
                if (TRA) {
                    if (gk     < K) v0 = A[(long long)gk * lda + gr];
                    if (gk + 1 < K) v1 = A[(long long)(gk + 1) * lda + gr];
                } else {
                    if (gk     < K) v0 = A[(long long)gr * lda + gk];
                    if (gk + 1 < K) v1 = A[(long long)gr * lda + gk + 1];
                }
            }
            h2 pk = { (_Float16)v0, (_Float16)v1 };
            *(h2*)(&As[r * APITCH + kperm(kp)]) = pk;
        }
        // stage B tile 32x64, stored N-major with permuted K
#pragma unroll
        for (int i = 0; i < 4; ++i) {
            int idx = tid + (i << 8);           // 1024 K-pairs
            int n   = idx & 63;
            int kp  = (idx >> 6) << 1;
            int gn = n0 + n, gk = kc + kp;
            float v0 = 0.f, v1 = 0.f;
            if (gn < N) {
                if (TRB) {
                    if (gk     < K) v0 = Bw[(long long)gn * ldb + gk];
                    if (gk + 1 < K) v1 = Bw[(long long)gn * ldb + gk + 1];
                } else {
                    if (gk     < K) v0 = Bw[(long long)gk * ldb + gn];
                    if (gk + 1 < K) v1 = Bw[(long long)(gk + 1) * ldb + gn];
                }
            }
            h2 pk = { (_Float16)v0, (_Float16)v1 };
            *(h2*)(&Bs[n * BPITCH + kperm(kp)]) = pk;
        }
        __syncthreads();

        v16h af = load_frag(As + (wid * 16 + (lane & 15)) * APITCH, lane);
#pragma unroll
        for (int j = 0; j < 4; ++j) {
            v16h bf = load_frag(Bs + (j * 16 + (lane & 15)) * BPITCH, lane);
            acc[j] = wmma_f16(af, bf, acc[j]);
        }
        __syncthreads();
    }

    // epilogue: VGPR r -> row r (lanes 0-15) / r+8 (lanes 16-31); lane&15 -> col
    const int rbase = m0 + wid * 16 + ((lane >> 4) << 3);
    const int c0 = lane & 15;
#pragma unroll
    for (int j = 0; j < 4; ++j) {
        int col = n0 + j * 16 + c0;
        if (col < N) {
#pragma unroll
            for (int r = 0; r < 8; ++r) {
                int row = rbase + r;
                if (row < M) {
                    float v = acc[j][r];
                    if (BIAS) v += bias[col];
                    if (RES)  v += R[(long long)row * ldc + col];
                    if (ACT == 1) v = fmaxf(v, 0.f);
                    C[(long long)row * ldc + col] = v;
                }
            }
        }
    }
}

// ---------------------------------------------------------------------------
// Fused sin-embedding(dist) + 128x128 linear (never materializes embeddings)
// grid: (ceil(PP/128), 2, B).  A rows built in swizzled fragment order.
// ---------------------------------------------------------------------------
__device__ __forceinline__ void emit_embed_row(_Float16* rowp, int half, float ind)
{
    for (int t = half * 32; t < half * 32 + 32; ++t) {
        float om = ind * __expf((float)(2 * t) * (-9.210340372f / 128.f));
        int k = 2 * t;
        int off = (k & ~31) + kperm(k & 31);
        h2 pk = { (_Float16)__sinf(om), (_Float16)__cosf(om) };
        *(h2*)(rowp + off) = pk;
    }
}

__global__ __launch_bounds__(256) void k_geo_dist(
    const float* __restrict__ pts, const float* __restrict__ W,
    const float* __restrict__ bias, float* __restrict__ out)
{
    __shared__ __align__(16) _Float16 As[128 * GPITCH]; // 128 pairs x K=128
    __shared__ __align__(16) _Float16 Bs[64 * BPITCH];

    const int b  = blockIdx.z;
    const int p0 = blockIdx.x * 128;
    const int n0 = blockIdx.y * 64;
    const float* P = pts + (long long)b * CNT * 3;
    const int tid = threadIdx.x, lane = tid & 31, wid = tid >> 5;

    {   // build A rows: 2 threads/row, 32 (sin,cos) pairs each
        int r = tid >> 1, half = tid & 1;
        int p = p0 + r;
        float ind = 0.f;
        if (p < CPP) {
            int i = p / CNT, j = p % CNT;
            float dx = P[i * 3]     - P[j * 3];
            float dy = P[i * 3 + 1] - P[j * 3 + 1];
            float dz = P[i * 3 + 2] - P[j * 3 + 2];
            float d2 = dx * dx + dy * dy + dz * dz;
            ind = sqrtf(fmaxf(d2, 0.f)) * 5.0f;           // /SIGMA_D (0.2)
        }
        emit_embed_row(&As[r * GPITCH], half, ind);
    }

    v8f acc[4] = {};
    for (int kc = 0; kc < 128; kc += 32) {
        __syncthreads();
#pragma unroll
        for (int i = 0; i < 4; ++i) {
            int idx = tid + (i << 8);
            int n   = idx & 63;
            int kp  = (idx >> 6) << 1;
            int gk = kc + kp;
            h2 pk = { (_Float16)W[(long long)gk * CHID + n0 + n],
                      (_Float16)W[(long long)(gk + 1) * CHID + n0 + n] };
            *(h2*)(&Bs[n * BPITCH + kperm(kp)]) = pk;
        }
        __syncthreads();
        v16h af = load_frag(As + (wid * 16 + (lane & 15)) * GPITCH + kc, lane);
#pragma unroll
        for (int j = 0; j < 4; ++j) {
            v16h bf = load_frag(Bs + (j * 16 + (lane & 15)) * BPITCH, lane);
            acc[j] = wmma_f16(af, bf, acc[j]);
        }
    }

    const int rbase = wid * 16 + ((lane >> 4) << 3);
    const int c0 = lane & 15;
#pragma unroll
    for (int j = 0; j < 4; ++j) {
        int col = n0 + j * 16 + c0;
#pragma unroll
        for (int r = 0; r < 8; ++r) {
            int p = p0 + rbase + r;
            if (p < CPP)
                out[((long long)b * CPP + p) * CHID + col] = acc[j][r] + bias[col];
        }
    }
}

// Fused sin-embedding(angle, 3 anchors) + linear + max over anchors, add into out
__global__ __launch_bounds__(256) void k_geo_angle(
    const float* __restrict__ pts, const int* __restrict__ knn,
    const float* __restrict__ W, const float* __restrict__ bias,
    float* __restrict__ out)
{
    __shared__ __align__(16) _Float16 As[128 * GPITCH];
    __shared__ __align__(16) _Float16 Bs[64 * BPITCH];

    const int b  = blockIdx.z;
    const int p0 = blockIdx.x * 128;
    const int n0 = blockIdx.y * 64;
    const float* P = pts + (long long)b * CNT * 3;
    const int tid = threadIdx.x, lane = tid & 31, wid = tid >> 5;

    v8f vmax[4];
#pragma unroll
    for (int j = 0; j < 4; ++j)
#pragma unroll
        for (int r = 0; r < 8; ++r) vmax[j][r] = -3.4e38f;

    for (int a = 0; a < 3; ++a) {
        __syncthreads();
        {   // build angle embeddings for this anchor
            int r = tid >> 1, half = tid & 1;
            int p = p0 + r;
            float ang = 0.f;
            if (p < CPP) {
                int i = p / CNT, j = p % CNT;
                const float* pi = P + i * 3;
                const float* pj = P + j * 3;
                int kn = knn[((long long)b * CNT + i) * 3 + a];
                const float* pk = P + kn * 3;
                float rx = pk[0] - pi[0], ry = pk[1] - pi[1], rz = pk[2] - pi[2];
                float ax = pj[0] - pi[0], ay = pj[1] - pi[1], az = pj[2] - pi[2];
                float cx = ry * az - rz * ay;
                float cy = rz * ax - rx * az;
                float cz = rx * ay - ry * ax;
                float sinv = sqrtf(cx * cx + cy * cy + cz * cz + 1e-12f);
                float cosv = rx * ax + ry * ay + rz * az;
                ang = atan2f(sinv, cosv) * 3.8197186342f;   // 180/(15*pi)
            }
            emit_embed_row(&As[r * GPITCH], half, ang);
        }
        v8f acc[4] = {};
        for (int kc = 0; kc < 128; kc += 32) {
            __syncthreads();
#pragma unroll
            for (int i = 0; i < 4; ++i) {
                int idx = tid + (i << 8);
                int n   = idx & 63;
                int kp  = (idx >> 6) << 1;
                int gk = kc + kp;
                h2 pk = { (_Float16)W[(long long)gk * CHID + n0 + n],
                          (_Float16)W[(long long)(gk + 1) * CHID + n0 + n] };
                *(h2*)(&Bs[n * BPITCH + kperm(kp)]) = pk;
            }
            __syncthreads();
            v16h af = load_frag(As + (wid * 16 + (lane & 15)) * GPITCH + kc, lane);
#pragma unroll
            for (int j = 0; j < 4; ++j) {
                v16h bf = load_frag(Bs + (j * 16 + (lane & 15)) * BPITCH, lane);
                acc[j] = wmma_f16(af, bf, acc[j]);
            }
        }
#pragma unroll
        for (int j = 0; j < 4; ++j)
#pragma unroll
            for (int r = 0; r < 8; ++r)
                vmax[j][r] = fmaxf(vmax[j][r], acc[j][r]);
    }

    const int rbase = wid * 16 + ((lane >> 4) << 3);
    const int c0 = lane & 15;
#pragma unroll
    for (int j = 0; j < 4; ++j) {
        int col = n0 + j * 16 + c0;
#pragma unroll
        for (int r = 0; r < 8; ++r) {
            int p = p0 + rbase + r;
            if (p < CPP)
                out[((long long)b * CPP + p) * CHID + col] += vmax[j][r] + bias[col];
        }
    }
}

// ---------------------------------------------------------------------------
// Farthest point sampling: one block per batch item
// ---------------------------------------------------------------------------
__global__ __launch_bounds__(256) void k_fps(const float* __restrict__ pts,
                                             int* __restrict__ idx,
                                             float* __restrict__ dists,
                                             int Npts, int nsample)
{
    const int b = blockIdx.x;
    pts   += (long long)b * Npts * 3;
    dists += (long long)b * Npts;
    idx   += (long long)b * nsample;

    __shared__ float smax[256];
    __shared__ int   simax[256];
    __shared__ int   s_last;
    const int tid = threadIdx.x;

    for (int i = tid; i < Npts; i += 256) dists[i] = 1e10f;
    if (tid == 0) { idx[0] = 0; s_last = 0; }
    __syncthreads();

    for (int it = 1; it < nsample; ++it) {
        int last = s_last;
        float lx = pts[last * 3], ly = pts[last * 3 + 1], lz = pts[last * 3 + 2];
        float bm = -1.f; int bi = 0;
        for (int i = tid; i < Npts; i += 256) {
            float dx = pts[i * 3] - lx, dy = pts[i * 3 + 1] - ly, dz = pts[i * 3 + 2] - lz;
            float d = dx * dx + dy * dy + dz * dz;
            float dm = fminf(dists[i], d);
            dists[i] = dm;
            if (dm > bm) { bm = dm; bi = i; }
        }
        smax[tid] = bm; simax[tid] = bi;
        __syncthreads();
        for (int s = 128; s > 0; s >>= 1) {
            if (tid < s && smax[tid + s] > smax[tid]) {
                smax[tid] = smax[tid + s]; simax[tid] = simax[tid + s];
            }
            __syncthreads();
        }
        if (tid == 0) { idx[it] = simax[0]; s_last = simax[0]; }
        __syncthreads();
    }
}

// ---------------------------------------------------------------------------
// Small helper kernels
// ---------------------------------------------------------------------------
__global__ void k_gather_pts(const float* __restrict__ dense,
                             const int* __restrict__ idx,
                             float* __restrict__ sp, int Bsz, int Ndense)
{
    int t = blockIdx.x * blockDim.x + threadIdx.x;
    if (t >= Bsz * CNT) return;
    int b = t / CNT, r = t % CNT;
    float x, y, z;
    if (r == 0) { x = y = z = 100.f; }
    else {
        int id = idx[b * CNP + (r - 1)];
        const float* p = dense + ((long long)b * Ndense + id) * 3;
        x = p[0]; y = p[1]; z = p[2];
    }
    sp[t * 3] = x; sp[t * 3 + 1] = y; sp[t * 3 + 2] = z;
}

__global__ void k_knn(const float* __restrict__ sp, int* __restrict__ knn, int Bsz)
{
    int t = blockIdx.x * blockDim.x + threadIdx.x;
    if (t >= Bsz * CNT) return;
    int b = t / CNT, i = t % CNT;
    const float* P = sp + (long long)b * CNT * 3;
    float bx = P[i * 3], by = P[i * 3 + 1], bz = P[i * 3 + 2];
    float bd[4] = {1e30f, 1e30f, 1e30f, 1e30f};
    int   bi[4] = {0, 0, 0, 0};
    for (int j = 0; j < CNT; ++j) {
        float dx = P[j * 3] - bx, dy = P[j * 3 + 1] - by, dz = P[j * 3 + 2] - bz;
        float d = dx * dx + dy * dy + dz * dz;
        if (d < bd[3]) {
            int p = 3;
            while (p > 0 && d < bd[p - 1]) { bd[p] = bd[p - 1]; bi[p] = bi[p - 1]; --p; }
            bd[p] = d; bi[p] = j;
        }
    }
    knn[t * 3] = bi[1]; knn[t * 3 + 1] = bi[2]; knn[t * 3 + 2] = bi[3];
}

__global__ void k_gather_feats(const float* __restrict__ f,
                               const float* __restrict__ bg,
                               const int* __restrict__ idx,
                               float* __restrict__ s, int Bsz, int Ndense)
{
    long long t = (long long)blockIdx.x * blockDim.x + threadIdx.x;
    if (t >= (long long)Bsz * CNT * CHID) return;
    int c = (int)(t & (CHID - 1));
    long long r = t >> 7;
    int b = (int)(r / CNT), rr = (int)(r % CNT);
    float v = (rr == 0) ? bg[c]
            : f[((long long)b * Ndense + idx[b * CNP + rr - 1]) * CHID + c];
    s[t] = v;
}

// y = LN(x (+res)) * g + b ; one wave per 128-wide row
__global__ void k_layernorm(const float* __restrict__ x, const float* __restrict__ res,
                            const float* __restrict__ g, const float* __restrict__ b,
                            float* __restrict__ y, int M)
{
    int row  = blockIdx.x * (blockDim.x >> 5) + (threadIdx.x >> 5);
    int lane = threadIdx.x & 31;
    if (row >= M) return;
    const float* xr = x + (long long)row * CHID;
    const float* rr = res ? res + (long long)row * CHID : nullptr;
    float v[4]; float s = 0.f;
#pragma unroll
    for (int i = 0; i < 4; ++i) {
        int c = lane + 32 * i;
        v[i] = xr[c] + (rr ? rr[c] : 0.f);
        s += v[i];
    }
    for (int o = 16; o; o >>= 1) s += __shfl_xor(s, o, 32);
    float mean = s * (1.f / CHID);
    float vs = 0.f;
#pragma unroll
    for (int i = 0; i < 4; ++i) { float d = v[i] - mean; vs += d * d; }
    for (int o = 16; o; o >>= 1) vs += __shfl_xor(vs, o, 32);
    float rstd = rsqrtf(vs * (1.f / CHID) + 1e-5f);
#pragma unroll
    for (int i = 0; i < 4; ++i) {
        int c = lane + 32 * i;
        y[(long long)row * CHID + c] = (v[i] - mean) * rstd * g[c] + b[c];
    }
}

// softmax over rows of length L with pre-scale; one wave per row
__global__ void k_softmax(float* __restrict__ s, int rows, int L, float scale)
{
    int row  = blockIdx.x * (blockDim.x >> 5) + (threadIdx.x >> 5);
    int lane = threadIdx.x & 31;
    if (row >= rows) return;
    float* p = s + (long long)row * L;
    float mx = -1e30f;
    for (int i = lane; i < L; i += 32) mx = fmaxf(mx, p[i] * scale);
    for (int o = 16; o; o >>= 1) mx = fmaxf(mx, __shfl_xor(mx, o, 32));
    float sum = 0.f;
    for (int i = lane; i < L; i += 32) {
        float e = __expf(p[i] * scale - mx);
        p[i] = e; sum += e;
    }
    for (int o = 16; o; o >>= 1) sum += __shfl_xor(sum, o, 32);
    float inv = 1.f / sum;
    for (int i = lane; i < L; i += 32) p[i] *= inv;
}

// scores[b,h,n,m] += dot_{c<32}( q[b,n,h*32+c], pe[(b,n,m),h*32+c] )
__global__ void k_qpe(const float* __restrict__ q, const float* __restrict__ pe,
                      float* __restrict__ scores, int Bsz)
{
    long long idx = (long long)blockIdx.x * blockDim.x + threadIdx.x;
    long long tot = (long long)Bsz * CHEADS * CNT * CNT;
    if (idx >= tot) return;
    int m = (int)(idx % CNT);
    long long t = idx / CNT;
    int n = (int)(t % CNT); t /= CNT;
    int h = (int)(t % CHEADS);
    int b = (int)(t / CHEADS);
    const float* qp = q  + ((long long)b * CNT + n) * CHID + h * 32;
    const float* pp = pe + (((long long)b * CNT + n) * CNT + m) * CHID + h * 32;
    float s = 0.f;
#pragma unroll
    for (int c = 0; c < 32; ++c) s += qp[c] * pp[c];
    scores[idx] += s;
}

// focused linear-attention feature map; one wave per row
__global__ void k_focus(const float* __restrict__ in, const float* __restrict__ scale,
                        float* __restrict__ out, int M)
{
    int row  = blockIdx.x * (blockDim.x >> 5) + (threadIdx.x >> 5);
    int lane = threadIdx.x & 31;
    if (row >= M) return;
    float v[4];
#pragma unroll
    for (int i = 0; i < 4; ++i) {
        int c = lane + 32 * i;
        float sp = log1pf(__expf(scale[c]));             // softplus
        v[i] = (fmaxf(in[(long long)row * CHID + c], 0.f) + 1e-6f) / sp;
    }
    float s2 = 0.f;
#pragma unroll
    for (int i = 0; i < 4; ++i) s2 += v[i] * v[i];
    for (int o = 16; o; o >>= 1) s2 += __shfl_xor(s2, o, 32);
    float qn = sqrtf(s2);
    float w[4]; float s6 = 0.f;
#pragma unroll
    for (int i = 0; i < 4; ++i) { w[i] = v[i] * v[i] * v[i]; s6 += w[i] * w[i]; }
    for (int o = 16; o; o >>= 1) s6 += __shfl_xor(s6, o, 32);
    float f = qn / sqrtf(s6);
#pragma unroll
    for (int i = 0; i < 4; ++i) {
        int c = lane + 32 * i;
        out[(long long)row * CHID + c] = w[i] * f;
    }
}

__global__ void k_colsum(const float* __restrict__ kmat, float* __restrict__ out,
                         int rowsPerB)
{
    int b = blockIdx.x, c = threadIdx.x;
    float s = 0.f;
    for (int m = 0; m < rowsPerB; ++m)
        s += kmat[((long long)b * rowsPerB + m) * CHID + c];
    out[b * CHID + c] = s;
}

__global__ void k_zrow(const float* __restrict__ q, const float* __restrict__ ksum,
                       float* __restrict__ z, int M, int rowsPerB)
{
    int row  = blockIdx.x * (blockDim.x >> 5) + (threadIdx.x >> 5);
    int lane = threadIdx.x & 31;
    if (row >= M) return;
    int b = row / rowsPerB;
    float s = 0.f;
#pragma unroll
    for (int i = 0; i < 4; ++i) {
        int c = lane + 32 * i;
        s += q[(long long)row * CHID + c] * ksum[b * CHID + c];
    }
    for (int o = 16; o; o >>= 1) s += __shfl_xor(s, o, 32);
    if (lane == 0) z[row] = 1.f / (s + 1e-6f);
}

__global__ void k_rowscale(float* __restrict__ h, const float* __restrict__ z,
                           long long tot)
{
    long long t = (long long)blockIdx.x * blockDim.x + threadIdx.x;
    if (t >= tot) return;
    h[t] *= z[t >> 7];
}

// ---------------------------------------------------------------------------
// Host-side plumbing
// ---------------------------------------------------------------------------
struct AttnP {
    const float *qW, *qb, *kW, *kb, *vW, *vb, *oW, *ob;
    const float *eW, *eb, *sW, *sb, *n1g, *n1b, *n2g, *n2b;
    const float *pW, *pb;
};

static void launch_gemm(hipStream_t st, int act, int tra, int trb, int hasb, int hasr,
                        const float* A, const float* Bw, const float* bias,
                        const float* res, float* C, int M, int N, int K,
                        int lda, int ldb, int ldc,
                        long long sAb, long long sAh, long long sBb, long long sBh,
                        long long sCb, long long sCh, int Hn, int nz)
{
    dim3 g((M + 127) / 128, (N + 63) / 64, nz), blk(256);
#define GEMM_CASE(a, ta, tb, bb, rr)                                           \
    if (act == a && tra == ta && trb == tb && hasb == bb && hasr == rr) {      \
        k_gemm<a, ta, tb, bb, rr><<<g, blk, 0, st>>>(A, Bw, bias, res, C, M,   \
            N, K, lda, ldb, ldc, sAb, sAh, sBb, sBh, sCb, sCh, Hn);            \
        return;                                                                \
    }
    GEMM_CASE(0, 0, 0, 1, 0)
    GEMM_CASE(0, 0, 0, 1, 1)
    GEMM_CASE(1, 0, 0, 1, 0)
    GEMM_CASE(0, 0, 1, 0, 0)
    GEMM_CASE(0, 0, 0, 0, 0)
    GEMM_CASE(0, 1, 0, 0, 0)
#undef GEMM_CASE
}

static void lin(hipStream_t st, const float* x, const float* W, const float* b,
                float* y, int M, int N, int K)
{
    launch_gemm(st, 0, 0, 0, 1, 0, x, W, b, nullptr, y, M, N, K, K, N, N,
                0, 0, 0, 0, 0, 0, 1, 1);
}

struct Scratch {
    float *q_s, *k_s, *v_s, *sA, *sB, *e_s, *pe, *scores;
    float *bq, *bq2, *bh, *bt, *e_big, *kv, *ksum, *z;
    int Bsz;
};

static void ffn_ln(hipStream_t st, const AttnP& p, float* x, int M,
                   float* etmp, float* ttmp)
{
    launch_gemm(st, 1, 0, 0, 1, 0, x, p.eW, p.eb, nullptr, etmp, M, 256, 128,
                128, 256, 256, 0, 0, 0, 0, 0, 0, 1, 1);
    launch_gemm(st, 0, 0, 0, 1, 1, etmp, p.sW, p.sb, x, ttmp, M, 128, 256,
                256, 128, 128, 0, 0, 0, 0, 0, 0, 1, 1);
    k_layernorm<<<(M + 7) / 8, 256, 0, st>>>(ttmp, nullptr, p.n2g, p.n2b, x, M);
}

static void attn_core(hipStream_t st, const AttnP& p, float* x, const float* kvsrc,
                      const float* geo, const Scratch& sc, bool use_pe)
{
    const int B = sc.Bsz, M2 = B * CNT;
    lin(st, x,     p.qW, p.qb, sc.q_s, M2, 128, 128);
    lin(st, kvsrc, p.kW, p.kb, sc.k_s, M2, 128, 128);
    lin(st, kvsrc, p.vW, p.vb, sc.v_s, M2, 128, 128);
    // scores = q @ k^T  per (b,h); head h = column slice [32h,32h+32)
    launch_gemm(st, 0, 0, 1, 0, 0, sc.q_s, sc.k_s, nullptr, nullptr, sc.scores,
                CNT, CNT, 32, 128, 128, CNT,
                (long long)CNT * 128, 32, (long long)CNT * 128, 32,
                (long long)CHEADS * CPP, (long long)CPP, CHEADS, B * CHEADS);
    if (use_pe) {
        lin(st, geo, p.pW, p.pb, sc.pe, B * CPP, 128, 128);
        long long tot = (long long)B * CHEADS * CPP;
        k_qpe<<<(unsigned)((tot + 255) / 256), 256, 0, st>>>(sc.q_s, sc.pe,
                                                             sc.scores, B);
    }
    k_softmax<<<(B * CHEADS * CNT + 7) / 8, 256, 0, st>>>(
        sc.scores, B * CHEADS * CNT, CNT, 0.17677669529f);   // 1/sqrt(32)
    // h = attn @ v per (b,h) -> sA (B,CNT,128)
    launch_gemm(st, 0, 0, 0, 0, 0, sc.scores, sc.v_s, nullptr, nullptr, sc.sA,
                CNT, 32, CNT, CNT, 128, 128,
                (long long)CHEADS * CPP, (long long)CPP,
                (long long)CNT * 128, 32, (long long)CNT * 128, 32,
                CHEADS, B * CHEADS);
    // out proj + residual, LN, FFN
    launch_gemm(st, 0, 0, 0, 1, 1, sc.sA, p.oW, p.ob, x, sc.sB, M2, 128, 128,
                128, 128, 128, 0, 0, 0, 0, 0, 0, 1, 1);
    k_layernorm<<<(M2 + 7) / 8, 256, 0, st>>>(sc.sB, nullptr, p.n1g, p.n1b, x, M2);
    ffn_ln(st, p, x, M2, sc.e_s, sc.sB);
}

static void lin_attn(hipStream_t st, const AttnP& p, const float* lscale,
                     const float* xq, const float* mem, float* outx,
                     const Scratch& sc)
{
    const int B = sc.Bsz, M3 = B * DENSE_N, Mm = B * CNT;
    lin(st, xq,  p.qW, p.qb, sc.bq,  M3, 128, 128);
    lin(st, mem, p.kW, p.kb, sc.k_s, Mm, 128, 128);
    lin(st, mem, p.vW, p.vb, sc.v_s, Mm, 128, 128);
    k_focus<<<(M3 + 7) / 8, 256, 0, st>>>(sc.bq,  lscale, sc.bq2, M3);
    k_focus<<<(Mm + 7) / 8, 256, 0, st>>>(sc.k_s, lscale, sc.sA,  Mm);
    k_colsum<<<B, 128, 0, st>>>(sc.sA, sc.ksum, CNT);
    k_zrow<<<(M3 + 7) / 8, 256, 0, st>>>(sc.bq2, sc.ksum, sc.z, M3, DENSE_N);
    // kv = k^T @ v  per batch (128x128)
    launch_gemm(st, 0, 1, 0, 0, 0, sc.sA, sc.v_s, nullptr, nullptr, sc.kv,
                128, 128, CNT, 128, 128, 128,
                (long long)CNT * 128, 0, (long long)CNT * 128, 0,
                128LL * 128, 0, 1, B);
    // h = q @ kv  per batch
    launch_gemm(st, 0, 0, 0, 0, 0, sc.bq2, sc.kv, nullptr, nullptr, sc.bh,
                DENSE_N, 128, 128, 128, 128, 128,
                (long long)DENSE_N * 128, 0, 128LL * 128, 0,
                (long long)DENSE_N * 128, 0, 1, B);
    k_rowscale<<<(unsigned)(((long long)M3 * 128 + 255) / 256), 256, 0, st>>>(
        sc.bh, sc.z, (long long)M3 * 128);
    launch_gemm(st, 0, 0, 0, 1, 1, sc.bh, p.oW, p.ob, xq, sc.bt, M3, 128, 128,
                128, 128, 128, 0, 0, 0, 0, 0, 0, 1, 1);
    k_layernorm<<<(M3 + 7) / 8, 256, 0, st>>>(sc.bt, nullptr, p.n1g, p.n1b, outx, M3);
    ffn_ln(st, p, outx, M3, sc.e_big, sc.bt);
}

// params are flattened in sorted-key order (jax pytree):
// kinds: 0=cross(16), 1=lin(17), 2=self(18)
static AttnP mkattn(void* const* d, int base, int kind)
{
    AttnP p{};
    p.eW  = (const float*)d[base + 0];  p.eb  = (const float*)d[base + 1];
    p.kW  = (const float*)d[base + 2];  p.kb  = (const float*)d[base + 3];
    p.n1b = (const float*)d[base + 4];  p.n1g = (const float*)d[base + 5];
    p.n2b = (const float*)d[base + 6];  p.n2g = (const float*)d[base + 7];
    p.oW  = (const float*)d[base + 8];  p.ob  = (const float*)d[base + 9];
    int o = 10;
    if (kind == 2) { p.pW = (const float*)d[base + 10]; p.pb = (const float*)d[base + 11]; o = 12; }
    p.qW = (const float*)d[base + o + 0]; p.qb = (const float*)d[base + o + 1];
    p.sW = (const float*)d[base + o + 2]; p.sb = (const float*)d[base + o + 3];
    if (kind == 1) o += 1;  // skip 'scale'
    p.vW = (const float*)d[base + o + 4]; p.vb = (const float*)d[base + o + 5];
    return p;
}

extern "C" void kernel_launch(void* const* d_in, const int* in_sizes, int n_in,
                              void* d_out, int out_size, void* d_ws, size_t ws_size,
                              hipStream_t stream)
{
    const int Bsz = in_sizes[0] / (DENSE_N * CHID);

    const float* f1 = (const float*)d_in[0];
    const float* f2 = (const float*)d_in[1];
    const float* po = (const float*)d_in[2];
    const float* pm = (const float*)d_in[3];
    const float* bg = (const float*)d_in[4];

    AttnP cp = mkattn(d_in, 5, 0);   // blocks[0].cross
    AttnP lp = mkattn(d_in, 21, 1);  // blocks[0].lin
    const float* lin_scale = (const float*)d_in[35];
    AttnP sp = mkattn(d_in, 38, 2);  // blocks[0].self
    const float* geo_aW = (const float*)d_in[56];
    const float* geo_ab = (const float*)d_in[57];
    const float* geo_dW = (const float*)d_in[58];
    const float* geo_db = (const float*)d_in[59];
    const float* outW   = (const float*)d_in[60];
    const float* outb   = (const float*)d_in[61];

    // workspace carve
    char* w = (char*)d_ws;
    size_t off = 0;
    auto carve = [&](size_t bytes) -> void* {
        void* p = w + off;
        off = (off + bytes + 255) & ~(size_t)255;
        return p;
    };
    int*   idxm  = (int*)  carve((size_t)Bsz * CNP * 4);
    int*   idxo  = (int*)  carve((size_t)Bsz * CNP * 4);
    float* dists = (float*)carve((size_t)Bsz * DENSE_N * 4);
    float* spm   = (float*)carve((size_t)Bsz * CNT * 3 * 4);
    float* spo   = (float*)carve((size_t)Bsz * CNT * 3 * 4);
    int*   knnm  = (int*)  carve((size_t)Bsz * CNT * 3 * 4);
    int*   knno  = (int*)  carve((size_t)Bsz * CNT * 3 * 4);
    float* geom  = (float*)carve((size_t)Bsz * CPP * CHID * 4);
    float* geoo  = (float*)carve((size_t)Bsz * CPP * CHID * 4);
    float* s1    = (float*)carve((size_t)Bsz * CNT * CHID * 4);
    float* s2    = (float*)carve((size_t)Bsz * CNT * CHID * 4);
    float* n1    = (float*)carve((size_t)Bsz * DENSE_N * CHID * 4);
    float* n2    = (float*)carve((size_t)Bsz * DENSE_N * CHID * 4);

    Scratch sc{};
    sc.Bsz    = Bsz;
    sc.q_s    = (float*)carve((size_t)Bsz * CNT * CHID * 4);
    sc.k_s    = (float*)carve((size_t)Bsz * CNT * CHID * 4);
    sc.v_s    = (float*)carve((size_t)Bsz * CNT * CHID * 4);
    sc.sA     = (float*)carve((size_t)Bsz * CNT * CHID * 4);
    sc.sB     = (float*)carve((size_t)Bsz * CNT * CHID * 4);
    sc.e_s    = (float*)carve((size_t)Bsz * CNT * 256 * 4);
    sc.pe     = (float*)carve((size_t)Bsz * CPP * CHID * 4);
    sc.scores = (float*)carve((size_t)Bsz * CHEADS * CPP * 4);
    sc.bq     = (float*)carve((size_t)Bsz * DENSE_N * CHID * 4);
    sc.bq2    = (float*)carve((size_t)Bsz * DENSE_N * CHID * 4);
    sc.bh     = (float*)carve((size_t)Bsz * DENSE_N * CHID * 4);
    sc.bt     = (float*)carve((size_t)Bsz * DENSE_N * CHID * 4);
    sc.e_big  = (float*)carve((size_t)Bsz * DENSE_N * 256 * 4);
    sc.kv     = (float*)carve((size_t)Bsz * CHID * CHID * 4);
    sc.ksum   = (float*)carve((size_t)Bsz * CHID * 4);
    sc.z      = (float*)carve((size_t)Bsz * DENSE_N * 4);

    // ---- FPS + gathers + kNN -------------------------------------------------
    k_fps<<<Bsz, 256, 0, stream>>>(pm, idxm, dists, DENSE_N, CNP);
    k_fps<<<Bsz, 256, 0, stream>>>(po, idxo, dists, DENSE_N, CNP);
    {
        int tot = Bsz * CNT;
        k_gather_pts<<<(tot + 255) / 256, 256, 0, stream>>>(pm, idxm, spm, Bsz, DENSE_N);
        k_gather_pts<<<(tot + 255) / 256, 256, 0, stream>>>(po, idxo, spo, Bsz, DENSE_N);
        k_knn<<<(tot + 255) / 256, 256, 0, stream>>>(spm, knnm, Bsz);
        k_knn<<<(tot + 255) / 256, 256, 0, stream>>>(spo, knno, Bsz);
    }

    // ---- fused geo embeddings (sin-embed generated inside the WMMA A-tile) ---
    {
        dim3 g((CPP + 127) / 128, 2, Bsz), blk(256);
        k_geo_dist <<<g, blk, 0, stream>>>(spm, geo_dW, geo_db, geom);
        k_geo_angle<<<g, blk, 0, stream>>>(spm, knnm, geo_aW, geo_ab, geom);
        k_geo_dist <<<g, blk, 0, stream>>>(spo, geo_dW, geo_db, geoo);
        k_geo_angle<<<g, blk, 0, stream>>>(spo, knno, geo_aW, geo_ab, geoo);
    }

    // ---- sparse features -----------------------------------------------------
    {
        long long tot = (long long)Bsz * CNT * CHID;
        k_gather_feats<<<(unsigned)((tot + 255) / 256), 256, 0, stream>>>(
            f1, bg, idxm, s1, Bsz, DENSE_N);
        k_gather_feats<<<(unsigned)((tot + 255) / 256), 256, 0, stream>>>(
            f2, bg, idxo, s2, Bsz, DENSE_N);
    }

    // ---- s2d block -----------------------------------------------------------
    attn_core(stream, sp, s1, s1, geom, sc, true);   // RPE self on s1
    attn_core(stream, sp, s2, s2, geoo, sc, true);   // RPE self on s2
    attn_core(stream, cp, s1, s2, nullptr, sc, false); // cross: s1 <- mem s2
    attn_core(stream, cp, s2, s1, nullptr, sc, false); // cross: s2 <- mem s1'
    lin_attn(stream, lp, lin_scale, f1, s1, n1, sc);
    lin_attn(stream, lp, lin_scale, f2, s2, n2, sc);

    // ---- output projections --------------------------------------------------
    float* o1 = (float*)d_out;
    float* o2 = (float*)d_out + (size_t)Bsz * DENSE_N * CHID;
    lin(stream, n1, outW, outb, o1, Bsz * DENSE_N, 128, 128);
    lin(stream, n2, outW, outb, o2, Bsz * DENSE_N, 128, 128);
}
